// MindSpeedTEGroupedLinear_37039797960928
// MI455X (gfx1250) — matrix operand order repaired and checked by reading, not compiled, not run on hardware
//
#include <hip/hip_runtime.h>

// CDNA5 / gfx1250 grouped GEMM (MoE column-parallel linear), fp32 in/out,
// bf16 WMMA compute with fp32 accumulation, double-buffered LDS pipeline.
//
//   out[rows_of_e, n] = sum_k x[rows_of_e, k] * W[e, n, k]
//
// Tile: 128x128 per block, BK=32 (one v_wmma_f32_16x16x32_bf16 K-step).
// 256 threads = 8 wave32; each wave computes a 32x64 sub-tile (2x4 WMMA accums).

typedef __attribute__((ext_vector_type(16))) __bf16 v16bf;
typedef __attribute__((ext_vector_type(8)))  float  v8f;

#define BK      32
#define LDS_W   20   // uints per LDS row: 16 data + 4 pad (keeps 16B align, spreads banks)

union Frag {
    v16bf bf;
    uint4 u4[2];
};

struct Slab {            // one 128x32 fp32 slab's worth of per-thread data
    float4 a[4];         // x rows
    float4 b[4];         // W rows
};

// Pack two fp32 -> two bf16 (round to nearest, half-up) in 3 VALU ops:
// two v_add_nc_u32 + one v_perm_b32 selecting the high halves.
__device__ __forceinline__ unsigned int pack2_bf16(float lo, float hi) {
    unsigned int a = __float_as_uint(lo) + 0x8000u;
    unsigned int b = __float_as_uint(hi) + 0x8000u;
    return __builtin_amdgcn_perm(b, a, 0x07060302u);   // {b[31:16], a[31:16]}
}

__global__ __launch_bounds__(256)
void grouped_gemm_bf16_wmma(const float* __restrict__ X,        // [M, K]
                            const float* __restrict__ Wt,       // [E, N, K]
                            const int*   __restrict__ m_splits, // [E], multiples of 128
                            float* __restrict__ Out,            // [M, N]
                            int K, int N, int E) {
    __shared__ __align__(16) unsigned int lA[2][128][LDS_W];  // bf16 pairs, [m][k]
    __shared__ __align__(16) unsigned int lB[2][128][LDS_W];  // bf16 pairs, [n][k]

    const int nT = blockIdx.x;   // inner: 11 n-tiles -> expert W stays hot in L2
    const int mT = blockIdx.y;   // outer: walks experts in order

    // Map this m-tile to its expert (uniform scalar scan; tiles align to splits).
    int e = 0;
    {
        int acc = 0;
        for (int i = 0; i < E; ++i) {
            int tiles = m_splits[i] >> 7;   // split / 128
            if (mT < acc + tiles) { e = i; break; }
            acc += tiles;
        }
    }

    const float* Ag = X  + (size_t)mT * 128 * K;
    const float* Bg = Wt + (size_t)e * N * K + (size_t)nT * 128 * K;

    const int t    = threadIdx.x;
    const int lane = t & 31;
    const int wave = t >> 5;            // 0..7
    const int wM   = (wave & 3) * 32;   // wave row origin in tile
    const int wN   = (wave >> 2) * 64;  // wave col origin in tile

    // Staging assignment: thread -> (row 0..127, k-half 0/16) of the 128x32 slab.
    const int sr = t >> 1;
    const int sk = (t & 1) << 4;        // element offset 0 or 16
    const int su = sk >> 1;             // uint offset 0 or 8

    const int lrow = lane & 15;
    const int kqA  = (lane >> 4) << 2;  // A frag: uint base 0 or 4
    const int kqB  = (lane >> 4) << 2;  // B frag: uint base 0 or 4

    v8f acc[2][4];
    #pragma unroll
    for (int i = 0; i < 2; ++i)
        #pragma unroll
        for (int j = 0; j < 4; ++j)
            acc[i][j] = v8f{};

    auto loadSlab = [&](int kk) {
        Slab s;
        const float4* ap = (const float4*)(Ag + (size_t)sr * K + kk + sk);
        const float4* bp = (const float4*)(Bg + (size_t)sr * K + kk + sk);
        #pragma unroll
        for (int i = 0; i < 4; ++i) s.a[i] = ap[i];
        #pragma unroll
        for (int i = 0; i < 4; ++i) s.b[i] = bp[i];
        return s;
    };

    auto storeSlab = [&](const Slab& s, int buf) {
        uint4 pa0 = { pack2_bf16(s.a[0].x, s.a[0].y), pack2_bf16(s.a[0].z, s.a[0].w),
                      pack2_bf16(s.a[1].x, s.a[1].y), pack2_bf16(s.a[1].z, s.a[1].w) };
        uint4 pa1 = { pack2_bf16(s.a[2].x, s.a[2].y), pack2_bf16(s.a[2].z, s.a[2].w),
                      pack2_bf16(s.a[3].x, s.a[3].y), pack2_bf16(s.a[3].z, s.a[3].w) };
        uint4 pb0 = { pack2_bf16(s.b[0].x, s.b[0].y), pack2_bf16(s.b[0].z, s.b[0].w),
                      pack2_bf16(s.b[1].x, s.b[1].y), pack2_bf16(s.b[1].z, s.b[1].w) };
        uint4 pb1 = { pack2_bf16(s.b[2].x, s.b[2].y), pack2_bf16(s.b[2].z, s.b[2].w),
                      pack2_bf16(s.b[3].x, s.b[3].y), pack2_bf16(s.b[3].z, s.b[3].w) };
        *(uint4*)&lA[buf][sr][su]     = pa0;
        *(uint4*)&lA[buf][sr][su + 4] = pa1;
        *(uint4*)&lB[buf][sr][su]     = pb0;
        *(uint4*)&lB[buf][sr][su + 4] = pb1;
    };

    auto compute = [&](int buf) {
        // A 16x32 bf16: lane = M (mod 16); lane<16 -> K {0..7}+{16..23},
        //               lane>=16 -> K {8..15}+{24..31}.  [M][K]-contiguous LDS.
        Frag fa[2], fb[4];
        #pragma unroll
        for (int i = 0; i < 2; ++i) {
            const unsigned int* p = &lA[buf][wM + i * 16 + lrow][kqA];
            fa[i].u4[0] = *(const uint4*)p;
            fa[i].u4[1] = *(const uint4*)(p + 8);
        }
        // B 32x16 bf16: lane = N (mod 16); lane<16 -> K {0..7}+{16..23}, lane>=16 -> K {8..15}+{24..31}.
        // [N][K]-contiguous LDS (W is natively [out][in] row-major).
        #pragma unroll
        for (int j = 0; j < 4; ++j) {
            const unsigned int* p = &lB[buf][wN + j * 16 + lrow][kqB];
            fb[j].u4[0] = *(const uint4*)p;
            fb[j].u4[1] = *(const uint4*)(p + 8);
        }
        #pragma unroll
        for (int i = 0; i < 2; ++i)
            #pragma unroll
            for (int j = 0; j < 4; ++j)
                acc[i][j] = __builtin_amdgcn_wmma_f32_16x16x32_bf16(
                    /*neg_a=*/false, fa[i].bf,
                    /*neg_b=*/false, fb[j].bf,
                    /*c_mod=*/(short)0, acc[i][j],
                    /*reuse_a=*/false, /*reuse_b=*/false);
    };

    // ---- pipelined main loop: load slab k+1, compute slab k, store slab k+1 ----
    storeSlab(loadSlab(0), 0);
    __syncthreads();

    int cur = 0;
    for (int k0 = 0; k0 < K - BK; k0 += BK) {
        Slab s = loadSlab(k0 + BK);   // global loads issued before WMMA block
        compute(cur);
        storeSlab(s, cur ^ 1);        // fills the other buffer
        __syncthreads();              // one barrier per K-step
        cur ^= 1;
    }
    compute(cur);                     // epilogue: last slab

    // ---- writeback: C/D layout: lanes 0-15 -> M=r, N=lane; lanes 16-31 -> M=8+r ----
    const int r0g = mT * 128 + wM + ((lane >> 4) << 3);
    const int c0g = nT * 128 + wN + (lane & 15);
    #pragma unroll
    for (int i = 0; i < 2; ++i) {
        #pragma unroll
        for (int j = 0; j < 4; ++j) {
            float* op = Out + (size_t)(r0g + i * 16) * N + (c0g + j * 16);
            #pragma unroll
            for (int r = 0; r < 8; ++r)
                op[(size_t)r * N] = acc[i][j][r];
        }
    }
}

extern "C" void kernel_launch(void* const* d_in, const int* in_sizes, int n_in,
                              void* d_out, int out_size, void* d_ws, size_t ws_size,
                              hipStream_t stream) {
    const float* x      = (const float*)d_in[0];   // [M, K] fp32
    const float* W      = (const float*)d_in[1];   // [E, N, K] fp32
    const int*   splits = (const int*)d_in[2];     // [E] int32
    float*       out    = (float*)d_out;           // [M, N] fp32

    const int E = in_sizes[2];
    const int K = 2048;                 // hidden size (reference IN_SIZE)
    const int M = in_sizes[0] / K;      // total tokens
    const int N = (int)((long long)in_sizes[1] / ((long long)E * K)); // per-expert out

    dim3 grid(N / 128, M / 128);        // (11, 256): n-tiles inner for W reuse in L2
    grouped_gemm_bf16_wmma<<<grid, 256, 0, stream>>>(x, W, splits, out, K, N, E);
}
